// GaugeInvariantAttention_2121713844870
// MI455X (gfx1250) — matrix operand order repaired and checked by reading, not compiled, not run on hardware
//
#include <hip/hip_runtime.h>
#include <hip/hip_bf16.h>
#include <math.h>

// ---------------------------------------------------------------------------
// Gauge-invariant attention for MI455X (gfx1250), wave32 + WMMA f16->f32.
// B=2, S=4096, D=768, H=12, dk=64.  Output = [out (f32), attn_weights (f32)].
// ---------------------------------------------------------------------------

typedef __attribute__((ext_vector_type(16))) _Float16 v16h;
typedef __attribute__((ext_vector_type(8)))  _Float16 v8h;
typedef __attribute__((ext_vector_type(8)))  float    v8f;

#define DM    768
#define NH    12
#define DK    64
#define SLEN  4096
#define BATCH 2
#define ROWS  (BATCH * SLEN)   // 8192 flattened (b,s) rows

static __device__ __forceinline__ v8f wmma_f16(v16h a, v16h b, v8f c) {
  // D = A(16x32 f16) * B(32x16 f16) + C(16x16 f32)
  return __builtin_amdgcn_wmma_f32_16x16x32_f16(false, a, false, b,
                                                (short)0, c, false, false);
}

static __device__ __forceinline__ v8f vzero8() {
  v8f z;
#pragma unroll
  for (int i = 0; i < 8; ++i) z[i] = 0.0f;
  return z;
}

// A operand: 16(M) x 32(K) f16 tile from row-major memory [.., ld], rows
// row0..row0+15, K columns k0..k0+31.
// ISA A layout: lane<16 -> row=lane, K = 0..7 (elems 0..7), 16..23 (elems 8..15)
//               lane>=16 -> row=lane-16, K = 8..15, 24..31
static __device__ __forceinline__ v16h load_a(const _Float16* __restrict__ base,
                                              int ld, int row0, int k0) {
  const int lane = threadIdx.x & 31;
  const int hf   = lane >> 4;
  const _Float16* p = base + (size_t)(row0 + (lane & 15)) * ld + (k0 + hf * 8);
  v8h lo = *(const v8h*)p;          // K = k0 + hf*8 + [0..7]
  v8h hi = *(const v8h*)(p + 16);   // K = k0 + hf*8 + 16 + [0..7]
  v16h r;
#pragma unroll
  for (int i = 0; i < 8; ++i) { r[i] = lo[i]; r[8 + i] = hi[i]; }
  return r;
}

// B operand: 32(K) x 16(N) f16 tile. Memory layout: column-major over N,
// i.e. base[n][k] with k contiguous (ld elements per n-row).
// ISA B layout: lane<16 -> col=lane, K = 0..7, 16..23; lane>=16 -> K = 8..15, 24..31.
static __device__ __forceinline__ v16h load_b(const _Float16* __restrict__ base,
                                              int ld, int n0, int k0) {
  const int lane = threadIdx.x & 31;
  const _Float16* p = base + (size_t)(n0 + (lane & 15)) * ld
                           + (k0 + ((lane >> 4) << 3));
  v8h lo = *(const v8h*)p;
  v8h hi = *(const v8h*)(p + 16);
  v16h r;
#pragma unroll
  for (int i = 0; i < 8; ++i) { r[i] = lo[i]; r[8 + i] = hi[i]; }
  return r;
}

// ---------------------------------------------------------------------------
__global__ void cvt_f32_f16(const float* __restrict__ src,
                            _Float16* __restrict__ dst, int n) {
  int i = blockIdx.x * blockDim.x + threadIdx.x;
  const int stride = gridDim.x * blockDim.x;
  for (; i < n; i += stride) dst[i] = (_Float16)src[i];
}

// ---------------------------------------------------------------------------
// Stage 1: QKV projection + gauge.  One wave = one (head, 16-row) tile.
//   qg[bh][s][dk] = (Q*cos - V*sin) / sqrt(dk)   (scale folded in)
//   kk[bh][s][dk] = K
//   vt[bh][dk][s] = V transposed (for PV B-operand)
// ---------------------------------------------------------------------------
__global__ __launch_bounds__(256)
void qkv_proj_kernel(const _Float16* __restrict__ x16,
                     const _Float16* __restrict__ wq,
                     const _Float16* __restrict__ wk,
                     const _Float16* __restrict__ wv,
                     const float* __restrict__ phase,
                     _Float16* __restrict__ qg,
                     _Float16* __restrict__ kk,
                     _Float16* __restrict__ vt) {
  const int wave = (int)((blockIdx.x * blockDim.x + threadIdx.x) >> 5);
  const int h  = wave / (ROWS / 16);
  const int st = wave % (ROWS / 16);
  const int r0 = st * 16;                 // flat (b,s) row
  const int b  = r0 >> 12;                // r0 / 4096
  const int s0 = r0 & (SLEN - 1);
  const int lane = threadIdx.x & 31;
  const int hf = lane >> 4;
  const int ln = lane & 15;

  const float ph = phase[h];
  const float cs = __cosf(ph) * 0.125f;   // 1/sqrt(64) folded into Qg
  const float sn = __sinf(ph) * 0.125f;

  v8f aK[4], aV[4], aQ[4];
#pragma unroll
  for (int n = 0; n < 4; ++n) { aK[n] = vzero8(); aV[n] = vzero8(); aQ[n] = vzero8(); }

  const int e0 = h * DK;
  for (int k0 = 0; k0 < DM; k0 += 32) {
    const v16h ax = load_a(x16, DM, r0, k0);   // x rows, K = feature dim
#pragma unroll
    for (int n = 0; n < 4; ++n) {
      aK[n] = wmma_f16(ax, load_b(wk, DM, e0 + n * 16, k0), aK[n]);
      aV[n] = wmma_f16(ax, load_b(wv, DM, e0 + n * 16, k0), aV[n]);
      aQ[n] = wmma_f16(ax, load_b(wq, DM, e0 + n * 16, k0), aQ[n]);
    }
  }

  const size_t bh = (size_t)b * NH + h;
  _Float16* kb = kk + bh * (size_t)SLEN * DK;
  _Float16* qb = qg + bh * (size_t)SLEN * DK;
  _Float16* vb = vt + bh * (size_t)DK * SLEN;

#pragma unroll
  for (int n = 0; n < 4; ++n) {
    const int d = n * 16 + ln;            // D-tile: col = lane&15
    v8h vpack;
#pragma unroll
    for (int j = 0; j < 8; ++j) {         // D-tile: row = j + hf*8
      const int s = s0 + hf * 8 + j;
      kb[(size_t)s * DK + d] = (_Float16)aK[n][j];
      qb[(size_t)s * DK + d] = (_Float16)(aQ[n][j] * cs - aV[n][j] * sn);
      vpack[j] = (_Float16)aV[n][j];      // same d, consecutive s -> pack
    }
    *(v8h*)(vb + (size_t)d * SLEN + (s0 + hf * 8)) = vpack;
  }
}

// ---------------------------------------------------------------------------
// Stage 2: attention.  One wave = one (b,h, 16-query) tile.
// Computes S^T = K @ Qg^T so that the query index lives on the lane axis:
//  - softmax stats are per-lane (+1 shfl_xor(16) to merge halves)
//  - the S^T D-tile IS the A-operand layout of P for the PV matmul.
// Pass 1: online (m, l).  Pass 2: recompute, write normalized p, accumulate PV.
// ---------------------------------------------------------------------------
__global__ __launch_bounds__(256)
void attn_kernel(const _Float16* __restrict__ qg,
                 const _Float16* __restrict__ kk,
                 const _Float16* __restrict__ vt,
                 float* __restrict__ attn,       // [B*NH][S][S]
                 _Float16* __restrict__ ctx) {   // [ROWS][DM] f16
  const int wave = (int)((blockIdx.x * blockDim.x + threadIdx.x) >> 5);
  const int bh = wave / (SLEN / 16);
  const int qt = wave % (SLEN / 16);
  const int q0 = qt * 16;
  const int lane = threadIdx.x & 31;
  const int hf = lane >> 4;
  const int ln = lane & 15;

  const _Float16* qb = qg + (size_t)bh * SLEN * DK;
  const _Float16* kb = kk + (size_t)bh * SLEN * DK;
  const _Float16* vb = vt + (size_t)bh * DK * SLEN;

  // Qg^T as B operand (columns = queries); invariant across the key loop.
  const v16h bq0 = load_b(qb, DK, q0, 0);
  const v16h bq1 = load_b(qb, DK, q0, 32);

  // ---- pass 1: row max m and denominator l (per lane = per query) ----
  float m = -1e30f, l = 0.0f;
  for (int t0 = 0; t0 < SLEN; t0 += 16) {
    if (t0 + 16 < SLEN)
      __builtin_prefetch(kb + (size_t)(t0 + 16 + ln) * DK, 0, 1);
    const v16h ak0 = load_a(kb, DK, t0, 0);
    const v16h ak1 = load_a(kb, DK, t0, 32);
    v8f c = vzero8();
    c = wmma_f16(ak0, bq0, c);
    c = wmma_f16(ak1, bq1, c);           // c[j] = score(t0+hf*8+j, q0+ln)
    float tmax = c[0];
#pragma unroll
    for (int i = 1; i < 8; ++i) tmax = fmaxf(tmax, c[i]);
    tmax = fmaxf(tmax, __shfl_xor(tmax, 16, 32));
    const float mn = fmaxf(m, tmax);
    const float alpha = __expf(m - mn);
    float ps = 0.0f;
#pragma unroll
    for (int i = 0; i < 8; ++i) ps += __expf(c[i] - mn);
    ps += __shfl_xor(ps, 16, 32);
    l = l * alpha + ps;
    m = mn;
  }
  const float rinv = 1.0f / l;

  // ---- pass 2: normalized weights out + context accumulation ----
  v8f cacc[4];
#pragma unroll
  for (int n = 0; n < 4; ++n) cacc[n] = vzero8();

  // each lane owns query row q0+ln; its 8 values are consecutive in t
  float* arow = attn + ((size_t)bh * SLEN + (q0 + ln)) * SLEN + hf * 8;

  for (int t0 = 0; t0 < SLEN; t0 += 32) {
    if (t0 + 32 < SLEN)
      __builtin_prefetch(kb + (size_t)(t0 + 32 + ln) * DK, 0, 1);
    v8f c0 = vzero8(), c1 = vzero8();
    {
      const v16h a0 = load_a(kb, DK, t0, 0);
      const v16h a1 = load_a(kb, DK, t0, 32);
      c0 = wmma_f16(a0, bq0, c0);
      c0 = wmma_f16(a1, bq1, c0);
    }
    {
      const v16h a0 = load_a(kb, DK, t0 + 16, 0);
      const v16h a1 = load_a(kb, DK, t0 + 16, 32);
      c1 = wmma_f16(a0, bq0, c1);
      c1 = wmma_f16(a1, bq1, c1);
    }
    v8f p0, p1;
    v16h pa;                       // P as A-operand [16q x 32t], zero shuffles
#pragma unroll
    for (int i = 0; i < 8; ++i) {
      p0[i] = __expf(c0[i] - m) * rinv;
      p1[i] = __expf(c1[i] - m) * rinv;
      pa[i]     = (_Float16)p0[i];
      pa[8 + i] = (_Float16)p1[i];
    }
    *(v8f*)(arow + t0)      = p0;  // contiguous 32B per lane
    *(v8f*)(arow + t0 + 16) = p1;
#pragma unroll
    for (int n = 0; n < 4; ++n)
      cacc[n] = wmma_f16(pa, load_b(vb, SLEN, n * 16, t0), cacc[n]);
  }

  // store context tile -> ctx16[b*SLEN + q][h*64 + d]
  const int b = bh / NH, h = bh % NH;
  _Float16* cb = ctx + (size_t)b * SLEN * DM;
#pragma unroll
  for (int n = 0; n < 4; ++n) {
    const int d = h * DK + n * 16 + ln;
#pragma unroll
    for (int j = 0; j < 8; ++j) {
      const int s = q0 + hf * 8 + j;
      cb[(size_t)s * DM + d] = (_Float16)cacc[n][j];
    }
  }
}

// ---------------------------------------------------------------------------
// Stage 3: output projection  out[s,e] = sum_d ctx[s,d] * Wo[e,d]  (f32 out)
// ---------------------------------------------------------------------------
__global__ __launch_bounds__(256)
void out_proj_kernel(const _Float16* __restrict__ ctx,
                     const _Float16* __restrict__ wo,
                     float* __restrict__ out) {
  const int wave = (int)((blockIdx.x * blockDim.x + threadIdx.x) >> 5);
  const int ec = wave / (ROWS / 16);
  const int st = wave % (ROWS / 16);
  const int r0 = st * 16;
  const int e0 = ec * 64;
  const int lane = threadIdx.x & 31;
  const int hf = lane >> 4;
  const int ln = lane & 15;

  v8f acc[4];
#pragma unroll
  for (int n = 0; n < 4; ++n) acc[n] = vzero8();

  for (int k0 = 0; k0 < DM; k0 += 32) {
    const v16h ax = load_a(ctx, DM, r0, k0);
#pragma unroll
    for (int n = 0; n < 4; ++n)
      acc[n] = wmma_f16(ax, load_b(wo, DM, e0 + n * 16, k0), acc[n]);
  }
#pragma unroll
  for (int n = 0; n < 4; ++n) {
    const int e = e0 + n * 16 + ln;
#pragma unroll
    for (int j = 0; j < 8; ++j)
      out[(size_t)(r0 + hf * 8 + j) * DM + e] = acc[n][j];
  }
}

// ---------------------------------------------------------------------------
extern "C" void kernel_launch(void* const* d_in, const int* in_sizes, int n_in,
                              void* d_out, int out_size, void* d_ws,
                              size_t ws_size, hipStream_t stream) {
  (void)in_sizes; (void)n_in; (void)out_size; (void)ws_size;

  const float* x   = (const float*)d_in[0];
  const float* Wq  = (const float*)d_in[1];
  const float* Wk  = (const float*)d_in[2];
  const float* Wv  = (const float*)d_in[3];
  const float* Wo  = (const float*)d_in[4];
  const float* gph = (const float*)d_in[5];

  float* out  = (float*)d_out;                       // [ROWS][DM]
  float* attn = out + (size_t)ROWS * DM;             // [B*NH][S][S]

  // workspace carve-up (all f16, all offsets 32B aligned)
  _Float16* ws  = (_Float16*)d_ws;
  const size_t N_X = (size_t)ROWS * DM;              // 6,291,456
  const size_t N_W = (size_t)DM * DM;                // 589,824
  const size_t N_H = (size_t)BATCH * NH * SLEN * DK; // 6,291,456
  _Float16* x16  = ws;
  _Float16* wq16 = x16  + N_X;
  _Float16* wk16 = wq16 + N_W;
  _Float16* wv16 = wk16 + N_W;
  _Float16* wo16 = wv16 + N_W;
  _Float16* qg16 = wo16 + N_W;
  _Float16* k16  = qg16 + N_H;
  _Float16* vt16 = k16  + N_H;
  _Float16* ctx16= vt16 + N_H;                       // total ~67.6 MB

  cvt_f32_f16<<<4096, 256, 0, stream>>>(x,  x16,  (int)N_X);
  cvt_f32_f16<<<1152, 256, 0, stream>>>(Wq, wq16, (int)N_W);
  cvt_f32_f16<<<1152, 256, 0, stream>>>(Wk, wk16, (int)N_W);
  cvt_f32_f16<<<1152, 256, 0, stream>>>(Wv, wv16, (int)N_W);
  cvt_f32_f16<<<1152, 256, 0, stream>>>(Wo, wo16, (int)N_W);

  // 6144 waves per stage, 8 waves (256 threads) per block -> 768 blocks
  qkv_proj_kernel<<<768, 256, 0, stream>>>(x16, wq16, wk16, wv16, gph,
                                           qg16, k16, vt16);
  attn_kernel<<<768, 256, 0, stream>>>(qg16, k16, vt16, attn, ctx16);
  out_proj_kernel<<<768, 256, 0, stream>>>(ctx16, wo16, out);
}
